// Seq2Seq_58385785421922
// MI455X (gfx1250) — matrix operand order repaired and checked by reading, not compiled, not run on hardware
//
#include <hip/hip_runtime.h>

typedef unsigned short u16;                                   // bf16 storage
typedef __bf16 bf16x16 __attribute__((ext_vector_type(16)));
typedef float  floatx8 __attribute__((ext_vector_type(8)));
typedef unsigned int u32x4 __attribute__((ext_vector_type(4)));
typedef int i32x8 __attribute__((ext_vector_type(8)));
typedef int i32x4 __attribute__((ext_vector_type(4)));

#define NN    32
#define LSEQ  64
#define EDIM  512
#define HDIM  1024
#define VOCAB 32000
#define KCH   32                 // k-chunk per pipeline stage
#define NCHUNK (HDIM / KCH)      // 32 stages in the output GEMM

#if __has_builtin(__builtin_amdgcn_tensor_load_to_lds) && __has_builtin(__builtin_amdgcn_s_wait_tensorcnt)
#define HAVE_TDM 1
#else
#define HAVE_TDM 0
#endif

// ---------------- device helpers ----------------

__device__ __forceinline__ u16 f2bf(float f) {
  unsigned u = __builtin_bit_cast(unsigned, f);
  u += 0x7fffu + ((u >> 16) & 1u);                            // round-nearest-even
  return (u16)(u >> 16);
}

// A fragment (16x32 bf16): lane half h holds K = k0+8h..+7 and K = k0+16+8h..+7
__device__ __forceinline__ bf16x16 load_a_frag(const u16* p) {
  union { bf16x16 v; uint4 q[2]; } t;
  t.q[0] = *reinterpret_cast<const uint4*>(p);
  t.q[1] = *reinterpret_cast<const uint4*>(p + 16);
  return t.v;
}

// B fragment (32x16 bf16): lane = column, lane half h holds K = k0+8h..+7 and K = k0+16+8h..+7
__device__ __forceinline__ bf16x16 load_b_frag(const u16* p) {
  union { bf16x16 v; uint4 q[2]; } t;
  t.q[0] = *reinterpret_cast<const uint4*>(p);
  t.q[1] = *reinterpret_cast<const uint4*>(p + 16);
  return t.v;
}

__device__ __forceinline__ floatx8 wmma_bf16(bf16x16 a, bf16x16 b, floatx8 c) {
  return __builtin_amdgcn_wmma_f32_16x16x32_bf16(false, a, false, b, (short)0, c,
                                                 false, false);
}

__device__ __forceinline__ float sigm(float x) { return 1.0f / (1.0f + __expf(-x)); }
__device__ __forceinline__ float tanh_fast(float x) {
  float e = __expf(-2.0f * fabsf(x));
  float t = (1.0f - e) / (1.0f + e);
  return copysignf(t, x);
}

__device__ __forceinline__ unsigned lds_off(const void* p) {
  return (unsigned)(size_t)p;      // LDS aperture: addr[31:0] is the LDS byte address
}

#if HAVE_TDM
// Issue one 2D TDM tile load: rows x KCH bf16, row stride = HDIM elements.
// D# per ISA ch.8: group0 = {count=1, lds_addr, global_addr, type=2},
// group1 = {data_size=1(2B), tensor dims = tile dims, tile0=KCH, tile1=rows, stride0=HDIM}.
__device__ __forceinline__ void tdm_load_2d(unsigned lds_addr, unsigned long long gaddr,
                                            unsigned rows) {
  u32x4 g0;
  g0.x = 1u;                                                  // count=1, user mode
  g0.y = lds_addr;                                            // [63:32] lds_addr
  g0.z = (unsigned)(gaddr & 0xffffffffu);                     // global_addr[31:0]
  g0.w = (unsigned)((gaddr >> 32) & 0x01ffffffu) | (2u << 30);// addr[56:32] | type=2
  i32x8 g1;
  g1[0] = (int)(1u << 16);                                    // wg_mask=0, data_size=1 (2B)
  g1[1] = (int)((KCH & 0xffffu) << 16);                       // tensor_dim0 lo16
  g1[2] = (int)((rows & 0xffffu) << 16);                      // tdim0 hi=0 | tensor_dim1 lo16
  g1[3] = (int)((KCH & 0xffffu) << 16);                       // tdim1 hi=0 | tile_dim0
  g1[4] = (int)(rows & 0xffffu);                              // tile_dim1 | tile_dim2=0
  g1[5] = (int)(HDIM);                                        // tensor_dim0_stride lo32
  g1[6] = 0;                                                  // stride0 hi | stride1 lo
  g1[7] = 0;
  i32x4 z4 = {};
  i32x8 z8 = {};
  // 6-arg toolchain form: (g0, g1, g2, g3, g4, cpol); tile_dim2=0 => groups 2+ unused
  __builtin_amdgcn_tensor_load_to_lds(g0, g1, z4, z4, z8, 0);
}
#endif

// ---------------- kernels ----------------

__global__ __launch_bounds__(256) void cvt_bf16_kernel(const float* __restrict__ src,
                                                       u16* __restrict__ dst, long n) {
  long i = (long)blockIdx.x * blockDim.x + threadIdx.x;
  long stride = (long)gridDim.x * blockDim.x;
  for (; i < n; i += stride) dst[i] = f2bf(src[i]);
}

__global__ __launch_bounds__(256) void gather_embed_kernel(const int* __restrict__ tok,  // NN x LSEQ
                                                           const float* __restrict__ emb, // V x E
                                                           u16* __restrict__ xs) {        // L x NN x E
  const int t = blockIdx.x, n = blockIdx.y;
  const int token = tok[n * LSEQ + t];
  const float* src = emb + (size_t)token * EDIM;
  u16* dst = xs + ((size_t)t * NN + n) * EDIM;
  for (int e = threadIdx.x; e < EDIM; e += blockDim.x) dst[e] = f2bf(src[e]);
}

__global__ __launch_bounds__(256) void zero_state_kernel(float* __restrict__ c,
                                                         u16* __restrict__ h) {
  const int i = blockIdx.x * blockDim.x + threadIdx.x;
  if (i < NN * HDIM) { c[i] = 0.0f; h[i] = 0; }
}

// One step: z = x@Wx.T + bx + h@Wh.T ; gate math ; update c ; write h (bf16).
// Block = 8 waves; block owns 64 hidden units (all 4 gates), grid = H/64 = 16.
__global__ __launch_bounds__(256) void lstm_step_kernel(
    const u16* __restrict__ x_bf,     // NN x EDIM  (this timestep)
    const u16* __restrict__ h_bf_in,  // NN x HDIM
    const u16* __restrict__ Wx_bf,    // 4H x EDIM
    const u16* __restrict__ Wh_bf,    // 4H x HDIM
    const float* __restrict__ bx,     // 4H
    float* __restrict__ c,            // NN x HDIM (in/out, per-lane ownership)
    u16* __restrict__ h_bf_out) {     // NN x HDIM
  const int tid  = threadIdx.x;
  const int wave = tid >> 5;
  const int lane = tid & 31;
  const int mt   = wave & 1;          // row tile: rows mt*16..+15 (batch)
  const int cg   = wave >> 1;         // 0..3: 16-unit column group inside slab
  const int slab = blockIdx.x;        // 0..15: 64 hidden units
  const int half = lane >> 4;
  const int lr   = lane & 15;
  const int ubase = slab * 64 + cg * 16;

  floatx8 acc[4] = {};                // one accumulator per gate

  // x @ Wx.T
  for (int k0 = 0; k0 < EDIM; k0 += KCH) {
    bf16x16 a = load_a_frag(x_bf + (size_t)(mt * 16 + lr) * EDIM + k0 + half * 8);
#pragma unroll
    for (int g = 0; g < 4; ++g) {
      bf16x16 b = load_b_frag(Wx_bf + (size_t)(g * HDIM + ubase + lr) * EDIM + k0 + half * 8);
      acc[g] = wmma_bf16(a, b, acc[g]);
    }
  }
  // h @ Wh.T
  for (int k0 = 0; k0 < HDIM; k0 += KCH) {
    bf16x16 a = load_a_frag(h_bf_in + (size_t)(mt * 16 + lr) * HDIM + k0 + half * 8);
#pragma unroll
    for (int g = 0; g < 4; ++g) {
      bf16x16 b = load_b_frag(Wh_bf + (size_t)(g * HDIM + ubase + lr) * HDIM + k0 + half * 8);
      acc[g] = wmma_bf16(a, b, acc[g]);
    }
  }

  const int u = ubase + lr;           // hidden unit (column of this lane)
  const float bi  = bx[u];
  const float bf_ = bx[HDIM + u];
  const float bg  = bx[2 * HDIM + u];
  const float bo_ = bx[3 * HDIM + u];
#pragma unroll
  for (int e = 0; e < 8; ++e) {
    const int row = mt * 16 + half * 8 + e;              // batch row
    const float zi = acc[0][e] + bi;
    const float zf = acc[1][e] + bf_;
    const float zg = acc[2][e] + bg;
    const float zo = acc[3][e] + bo_;
    const size_t idx = (size_t)row * HDIM + u;
    const float cn = sigm(zf) * c[idx] + sigm(zi) * tanh_fast(zg);
    c[idx] = cn;
    h_bf_out[idx] = f2bf(sigm(zo) * tanh_fast(cn));
  }
}

// logits = Hall @ Wo.T + bo ; Hall is (LT*NN) x HDIM row-major (row r = t*NN+n).
// Block tile 64 rows x 256 cols; 8 waves, each computes a 32x64 sub-block (2x4 WMMA tiles).
// A/B k-chunks staged into LDS by the Tensor Data Mover, triple-buffered on TENSORcnt
// (3 buffers => single barrier per stage: next stage's barrier orders the WAR reuse).
__global__ __launch_bounds__(256) void out_gemm_kernel(const u16* __restrict__ h_all,
                                                       const u16* __restrict__ Wo_bf, // V x H
                                                       const float* __restrict__ bo,
                                                       float* __restrict__ out) {    // NN x LT x V
  __shared__ alignas(16) u16 sA[3][64 * KCH];    //  4 KB per buffer
  __shared__ alignas(16) u16 sB[3][256 * KCH];   // 16 KB per buffer

  const int tid  = threadIdx.x;
  const int wave = tid >> 5;
  const int lane = tid & 31;
  const int mtw  = wave >> 2;                    // 0..1 : 32-row sub-block
  const int ntw  = wave & 3;                     // 0..3 : 64-col sub-block
  const int row0 = blockIdx.y * 64;
  const int col0 = blockIdx.x * 256;
  const int half = lane >> 4;
  const int lr   = lane & 15;

  floatx8 acc[2][4] = {};

#if HAVE_TDM
  const unsigned long long gA =
      (unsigned long long)(size_t)h_all + (size_t)row0 * HDIM * 2;
  const unsigned long long gB =
      (unsigned long long)(size_t)Wo_bf + (size_t)col0 * HDIM * 2;

  if (wave == 0) {                               // wave-uniform guard: only wave 0 drives TDM
#pragma unroll
    for (int i = 0; i < 2; ++i) {
      tdm_load_2d(lds_off(&sA[i][0]), gA + (size_t)i * KCH * 2, 64);
      tdm_load_2d(lds_off(&sB[i][0]), gB + (size_t)i * KCH * 2, 256);
    }
  }
#endif
  for (int i = 0; i < NCHUNK; ++i) {
    const int buf = i % 3;
#if HAVE_TDM
    if (wave == 0) {
      if (i + 1 < NCHUNK) __builtin_amdgcn_s_wait_tensorcnt(2);   // chunk i landed
      else                __builtin_amdgcn_s_wait_tensorcnt(0);   // last chunk landed
    }
    __syncthreads();   // chunk i visible to all waves; all waves done reading chunk i-1
    if (wave == 0 && i + 2 < NCHUNK) {
      const int nbuf = (i + 2) % 3;              // == (i-1)%3, already fully consumed
      tdm_load_2d(lds_off(&sA[nbuf][0]), gA + (size_t)(i + 2) * KCH * 2, 64);
      tdm_load_2d(lds_off(&sB[nbuf][0]), gB + (size_t)(i + 2) * KCH * 2, 256);
    }
#else
    {
      const int k0 = i * KCH;
      // cooperative fallback copy: A = 64x32 (256 uint4), B = 256x32 (1024 uint4)
      for (int j = tid; j < 256; j += 256) {
        const int r = j >> 2, p = j & 3;
        ((uint4*)&sA[buf][0])[j] =
            *(const uint4*)(h_all + (size_t)(row0 + r) * HDIM + k0 + p * 8);
      }
      for (int j = tid; j < 1024; j += 256) {
        const int r = j >> 2, p = j & 3;
        ((uint4*)&sB[buf][0])[j] =
            *(const uint4*)(Wo_bf + (size_t)(col0 + r) * HDIM + k0 + p * 8);
      }
    }
    __syncthreads();
#endif
    bf16x16 af[2], bfr[4];
#pragma unroll
    for (int m = 0; m < 2; ++m)
      af[m] = load_a_frag(&sA[buf][(mtw * 32 + m * 16 + lr) * KCH + half * 8]);
#pragma unroll
    for (int n = 0; n < 4; ++n)
      bfr[n] = load_b_frag(&sB[buf][(ntw * 64 + n * 16 + lr) * KCH + half * 8]);
#pragma unroll
    for (int m = 0; m < 2; ++m)
#pragma unroll
      for (int n = 0; n < 4; ++n)
        acc[m][n] = wmma_bf16(af[m], bfr[n], acc[m][n]);
#if !HAVE_TDM
    __syncthreads();                             // buffer reusable (fallback path only)
#endif
  }

#pragma unroll
  for (int m = 0; m < 2; ++m) {
    const int rbase = row0 + mtw * 32 + m * 16 + half * 8;
#pragma unroll
    for (int n = 0; n < 4; ++n) {
      const int v = col0 + ntw * 64 + n * 16 + lr;
      const float bias = bo[v];
#pragma unroll
      for (int e = 0; e < 8; ++e) {
        const int r = rbase + e;                 // r = t*NN + b
        const int t = r >> 5, b = r & 31;
        out[((size_t)b * LSEQ + t) * VOCAB + v] = acc[m][n][e] + bias;
      }
    }
  }
}

// In-place log-softmax over last dim; one block per (n,t) row.
__global__ __launch_bounds__(256) void logsoftmax_kernel(float* __restrict__ out) {
  float* p = out + (size_t)blockIdx.x * VOCAB;
  __shared__ float red[256];
  const int tid = threadIdx.x;

  float m = -3.4e38f;
  for (int i = tid; i < VOCAB; i += 256) m = fmaxf(m, p[i]);
  red[tid] = m; __syncthreads();
  for (int s = 128; s > 0; s >>= 1) {
    if (tid < s) red[tid] = fmaxf(red[tid], red[tid + s]);
    __syncthreads();
  }
  m = red[0]; __syncthreads();

  float sum = 0.0f;
  for (int i = tid; i < VOCAB; i += 256) sum += __expf(p[i] - m);
  red[tid] = sum; __syncthreads();
  for (int s = 128; s > 0; s >>= 1) {
    if (tid < s) red[tid] += red[tid + s];
    __syncthreads();
  }
  const float lse = m + __logf(red[0]);
  __syncthreads();
  for (int i = tid; i < VOCAB; i += 256) p[i] = p[i] - lse;
}

// ---------------- host launch ----------------

extern "C" void kernel_launch(void* const* d_in, const int* in_sizes, int n_in,
                              void* d_out, int out_size, void* d_ws, size_t ws_size,
                              hipStream_t stream) {
  (void)in_sizes; (void)n_in; (void)out_size; (void)ws_size;
  const int*   src     = (const int*)  d_in[0];
  const int*   tgt     = (const int*)  d_in[1];
  const float* emb_src = (const float*)d_in[2];
  const float* Wx_e    = (const float*)d_in[3];
  const float* bx_e    = (const float*)d_in[4];
  const float* Wh_e    = (const float*)d_in[5];
  const float* emb_tgt = (const float*)d_in[6];
  const float* Wx_d    = (const float*)d_in[7];
  const float* bx_d    = (const float*)d_in[8];
  const float* Wh_d    = (const float*)d_in[9];
  const float* Wo      = (const float*)d_in[10];
  const float* bo      = (const float*)d_in[11];
  float* out = (float*)d_out;

  char* ws = (char*)d_ws;
  size_t off = 0;
  auto alloc = [&](size_t bytes) -> void* {
    void* p = ws + off;
    off = (off + bytes + 255) & ~(size_t)255;
    return p;
  };

  u16* Wx_e_bf = (u16*)alloc((size_t)4 * HDIM * EDIM * 2);
  u16* Wh_e_bf = (u16*)alloc((size_t)4 * HDIM * HDIM * 2);
  u16* Wx_d_bf = (u16*)alloc((size_t)4 * HDIM * EDIM * 2);
  u16* Wh_d_bf = (u16*)alloc((size_t)4 * HDIM * HDIM * 2);
  u16* Wo_bf   = (u16*)alloc((size_t)VOCAB * HDIM * 2);
  u16* xs_e    = (u16*)alloc((size_t)LSEQ * NN * EDIM * 2);
  u16* xs_d    = (u16*)alloc((size_t)LSEQ * NN * EDIM * 2);
  u16* h_all   = (u16*)alloc((size_t)LSEQ * NN * HDIM * 2);  // decoder hidden states (bf16)
  u16* hA      = (u16*)alloc((size_t)NN * HDIM * 2);
  u16* hB      = (u16*)alloc((size_t)NN * HDIM * 2);
  float* cbuf  = (float*)alloc((size_t)NN * HDIM * 4);

  // 1) weights -> bf16
  cvt_bf16_kernel<<<2048, 256, 0, stream>>>(Wx_e, Wx_e_bf, (long)4 * HDIM * EDIM);
  cvt_bf16_kernel<<<2048, 256, 0, stream>>>(Wh_e, Wh_e_bf, (long)4 * HDIM * HDIM);
  cvt_bf16_kernel<<<2048, 256, 0, stream>>>(Wx_d, Wx_d_bf, (long)4 * HDIM * EDIM);
  cvt_bf16_kernel<<<2048, 256, 0, stream>>>(Wh_d, Wh_d_bf, (long)4 * HDIM * HDIM);
  cvt_bf16_kernel<<<4096, 256, 0, stream>>>(Wo,   Wo_bf,   (long)VOCAB * HDIM);

  // 2) embedding gathers -> bf16, [t][n][E]
  gather_embed_kernel<<<dim3(LSEQ, NN), 256, 0, stream>>>(src, emb_src, xs_e);
  gather_embed_kernel<<<dim3(LSEQ, NN), 256, 0, stream>>>(tgt, emb_tgt, xs_d);

  // 3) zero h0, c0 (deterministic every call)
  zero_state_kernel<<<(NN * HDIM + 255) / 256, 256, 0, stream>>>(cbuf, hA);

  // 4) encoder: 64 sequential steps, double-buffered h (cross-WG WAR)
  for (int t = 0; t < LSEQ; ++t) {
    const u16* hin  = (t & 1) ? hB : hA;
    u16*       hout = (t & 1) ? hA : hB;
    lstm_step_kernel<<<HDIM / 64, 256, 0, stream>>>(
        xs_e + (size_t)t * NN * EDIM, hin, Wx_e_bf, Wh_e_bf, bx_e, cbuf, hout);
  }
  const u16* h_enc_final = hA;  // LSEQ=64: last step (t=63, odd) wrote hA

  // 5) decoder: 64 sequential steps, each h_t stored into h_all for batched projection
  for (int t = 0; t < LSEQ; ++t) {
    const u16* hin  = (t == 0) ? h_enc_final : h_all + (size_t)(t - 1) * NN * HDIM;
    u16*       hout = h_all + (size_t)t * NN * HDIM;
    lstm_step_kernel<<<HDIM / 64, 256, 0, stream>>>(
        xs_d + (size_t)t * NN * EDIM, hin, Wx_d_bf, Wh_d_bf, bx_d, cbuf, hout);
  }

  // 6) batched output projection: (LT*NN=2048) x VOCAB, K=HDIM -> raw logits in d_out
  out_gemm_kernel<<<dim3(VOCAB / 256, (LSEQ * NN) / 64), 256, 0, stream>>>(
      h_all, Wo_bf, bo, out);

  // 7) in-place log-softmax per (n,t) row
  logsoftmax_kernel<<<LSEQ * NN, 256, 0, stream>>>(out);
}
